// PerceiverSelfAttention_33217277067713
// MI455X (gfx1250) — matrix operand order repaired and checked by reading, not compiled, not run on hardware
//
#include <hip/hip_runtime.h>
#include <hip/hip_bf16.h>
#include <stdint.h>

typedef _Float16 f16;
typedef _Float16 v16h __attribute__((ext_vector_type(16)));
typedef _Float16 v8h  __attribute__((ext_vector_type(8)));
typedef float    v8f  __attribute__((ext_vector_type(8)));

// Problem constants (match reference)
constexpr int Bz   = 4;
constexpr int LQ   = 512;
constexpr int LKV  = 8192;
constexpr int QDIM = 1024;
constexpr int KVDIM= 512;
constexpr int CH   = 256;   // QK_CH == V_CH
constexpr int H    = 8;
constexpr int DH   = 32;    // per-head dim (dqk == dv)

__device__ __forceinline__ v16h cat16(v8h lo, v8h hi) {
  return __builtin_shufflevector(lo, hi, 0,1,2,3,4,5,6,7,8,9,10,11,12,13,14,15);
}
__device__ __forceinline__ v8f zero8() {
  v8f z;
  #pragma unroll
  for (int i = 0; i < 8; ++i) z[i] = 0.0f;
  return z;
}
__device__ __forceinline__ v8f wmma_f16(v16h a, v16h b, v8f c) {
  // (neg_a, A, neg_b, B, c_mod, C, reuse_a, reuse_b)
  return __builtin_amdgcn_wmma_f32_16x16x32_f16(false, a, false, b, (short)0, c, false, false);
}
// Async DMA: 16 bytes global -> LDS, tracked by ASYNCcnt (no VGPR data return).
__device__ __forceinline__ void async_copy16(uint32_t lds_addr, const f16* gptr) {
  asm volatile("global_load_async_to_lds_b128 %0, %1, off"
               :: "v"(lds_addr), "v"(gptr) : "memory");
}

// ---------------------------------------------------------------------------
// LayerNorm (f32 -> f16), one block (256 threads = 8 waves) per row.
// ---------------------------------------------------------------------------
template <int COLS>
__global__ __launch_bounds__(256) void ln_to_f16(
    const float* __restrict__ x, const float* __restrict__ g,
    const float* __restrict__ bta, f16* __restrict__ y) {
  const int row = blockIdx.x;
  const float* xr = x + (size_t)row * COLS;
  float s = 0.f, ss = 0.f;
  for (int c = threadIdx.x; c < COLS; c += 256) {
    float v = xr[c]; s += v; ss += v * v;
  }
  #pragma unroll
  for (int m = 16; m >= 1; m >>= 1) {
    s  += __shfl_xor(s,  m);
    ss += __shfl_xor(ss, m);
  }
  __shared__ float sA[8], sB[8];
  const int wid = threadIdx.x >> 5;
  if ((threadIdx.x & 31) == 0) { sA[wid] = s; sB[wid] = ss; }
  __syncthreads();
  float ts = 0.f, tss = 0.f;
  #pragma unroll
  for (int i = 0; i < 8; ++i) { ts += sA[i]; tss += sB[i]; }
  const float mean = ts * (1.0f / COLS);
  const float var  = tss * (1.0f / COLS) - mean * mean;
  const float inv  = rsqrtf(var + 1e-5f);
  f16* yr = y + (size_t)row * COLS;
  for (int c = threadIdx.x; c < COLS; c += 256)
    yr[c] = (f16)((xr[c] - mean) * inv * g[c] + bta[c]);
}

// ---------------------------------------------------------------------------
// Cast W[K][N] f32 -> Wt[N][K] f16 (transpose so B-fragments are contiguous).
// ---------------------------------------------------------------------------
__global__ void cast_transpose(const float* __restrict__ W, f16* __restrict__ Wt,
                               int K, int N) {
  int idx = blockIdx.x * 256 + threadIdx.x;
  if (idx < K * N) {
    int k = idx / N, n = idx % N;
    Wt[(size_t)n * K + k] = (f16)W[idx];
  }
}

// ---------------------------------------------------------------------------
// WMMA GEMM: out = (A[M,K] @ Wt[N,K]^T + bias) * scale
// 256 threads = 8 waves, wave grid 4(m) x 2(n), block tile 64x32, 16x16x32 steps.
// out_mode 0: row-major f16 out[m*N + n]
// out_mode 1: V-transposed   out[(b*CH + n)*LKV + kv], m = b*LKV + kv
// ---------------------------------------------------------------------------
__global__ __launch_bounds__(256) void gemm_wmma_f16(
    const f16* __restrict__ A, const f16* __restrict__ Wt,
    const float* __restrict__ bias, f16* __restrict__ out,
    int M, int N, int K, float scale, int out_mode) {
  const int lane = threadIdx.x & 31;
  const int wave = threadIdx.x >> 5;
  const int wm = wave >> 1, wn = wave & 1;
  const int m0 = blockIdx.x * 64 + wm * 16;
  const int n0 = blockIdx.y * 32 + wn * 16;
  const int l16 = lane & 15, hh = lane >> 4;

  v8f acc = zero8();
  const f16* arow = A  + (size_t)(m0 + l16) * K;  // A-frag: lane holds row m=l16
  const f16* brow = Wt + (size_t)(n0 + l16) * K;  // B-frag: lane holds col n=l16

  for (int k0 = 0; k0 < K; k0 += 32) {
    v8h alo = *(const v8h*)(arow + k0 + hh * 8);        // K = hh*8 + e   (e=0..7)
    v8h ahi = *(const v8h*)(arow + k0 + 16 + hh * 8);   // K = 16+hh*8+e
    v8h blo = *(const v8h*)(brow + k0 + hh * 8);        // K = hh*8 + e   (e=0..7)
    v8h bhi = *(const v8h*)(brow + k0 + 16 + hh * 8);   // K = 16+hh*8+e
    acc = wmma_f16(cat16(alo, ahi), cat16(blo, bhi), acc);
  }

  const int n = n0 + l16;
  const float bb = bias[n];
  if (out_mode == 0) {
    #pragma unroll
    for (int r = 0; r < 8; ++r) {
      int m = m0 + hh * 8 + r;                          // C layout: M = hh*8 + r
      out[(size_t)m * N + n] = (f16)((acc[r] + bb) * scale);
    }
  } else {
    int mbase = m0 + hh * 8;
    int bidx = mbase / LKV, kv = mbase % LKV;           // 8 consecutive kv rows
    v8h pack;
    #pragma unroll
    for (int r = 0; r < 8; ++r) pack[r] = (f16)((acc[r] + bb) * scale);
    *(v8h*)(out + ((size_t)bidx * CH + n) * LKV + kv) = pack;
  }
}

// ---------------------------------------------------------------------------
// Flash attention: grid(B*H, LQ/128); 8 waves/block, one 16-row Q tile per wave.
// K/V tiles (32kv x 32d) staged into LDS with async global->LDS DMA, double
// buffered: wave 0 prefetches tile i+1 while all waves compute tile i.
// kv chunks of 32: 2 WMMAs (scores) + online softmax + 2 WMMAs (P@V).
// Q is pre-scaled by 1/sqrt(32) at Q-GEMM time.
// ---------------------------------------------------------------------------
__global__ __launch_bounds__(256) void flash_attn_wmma(
    const f16* __restrict__ Qh, const f16* __restrict__ Kh,
    const f16* __restrict__ Vt, float* __restrict__ out) {
  const int bh = blockIdx.x;
  const int b = bh >> 3, h = bh & 7;
  const int wave = threadIdx.x >> 5;
  const int lane = threadIdx.x & 31;
  const int l16 = lane & 15, hh = lane >> 4;
  const int q0 = blockIdx.y * 128 + wave * 16;

  __shared__ f16 Klds[2][32][32];  // [buf][kv_local][d]      (4 KB)
  __shared__ f16 Vlds[2][32][32];  // [buf][dv][kv_local]     (4 KB)
  __shared__ f16 Pl[8][16 * 32];   // per-wave P tile bounce  (8 KB)
  f16* pw = Pl[wave];

  // Q A-fragment (loaded once): lane m=l16
  const f16* qrow = Qh + ((size_t)(b * LQ + q0 + l16)) * CH + h * DH;
  const v16h qa = cat16(*(const v8h*)(qrow + hh * 8),
                        *(const v8h*)(qrow + 16 + hh * 8));

  // Async stage of one 32x32 K tile + 32x32 V^T tile: 8 x b128 per wave.
  auto stage = [&](int kv0, int buf) {
    const f16* kg = Kh + ((size_t)(b * LKV + kv0 + lane)) * CH + h * DH;
    const f16* vg = Vt + ((size_t)(b * CH + h * DH + lane)) * LKV + kv0;
    #pragma unroll
    for (int j = 0; j < 4; ++j) {
      async_copy16((uint32_t)(uintptr_t)&Klds[buf][lane][j * 8], kg + j * 8);
      async_copy16((uint32_t)(uintptr_t)&Vlds[buf][lane][j * 8], vg + j * 8);
    }
  };

  float rmax[8], rsum[8];
  #pragma unroll
  for (int r = 0; r < 8; ++r) { rmax[r] = -INFINITY; rsum[r] = 0.f; }
  v8f acc0 = zero8(), acc1 = zero8();

  if (wave == 0) {
    stage(0, 0);
    asm volatile("s_wait_asynccnt 0" ::: "memory");
  }
  __syncthreads();

  for (int kv0 = 0; kv0 < LKV; kv0 += 32) {
    const int buf = (kv0 >> 5) & 1;
    if (wave == 0 && kv0 + 32 < LKV) stage(kv0 + 32, buf ^ 1);

    // K B-fragments from LDS tile: lane n=l16, d = hh*8+e / 16+hh*8+e
    const f16* kl0 = &Klds[buf][l16][hh * 8];
    const f16* kl1 = &Klds[buf][16 + l16][hh * 8];
    v8f s0 = wmma_f16(qa, cat16(*(const v8h*)kl0, *(const v8h*)(kl0 + 16)), zero8());
    v8f s1 = wmma_f16(qa, cat16(*(const v8h*)kl1, *(const v8h*)(kl1 + 16)), zero8());

    float p0[8], p1[8];
    #pragma unroll
    for (int r = 0; r < 8; ++r) {
      float mx = fmaxf(s0[r], s1[r]);
      #pragma unroll
      for (int m = 1; m <= 8; m <<= 1) mx = fmaxf(mx, __shfl_xor(mx, m));
      const float mnew  = fmaxf(rmax[r], mx);
      const float alpha = __expf(rmax[r] - mnew);
      const float e0 = __expf(s0[r] - mnew);
      const float e1 = __expf(s1[r] - mnew);
      float ps = e0 + e1;
      #pragma unroll
      for (int m = 1; m <= 8; m <<= 1) ps += __shfl_xor(ps, m);
      rsum[r] = rsum[r] * alpha + ps;
      rmax[r] = mnew;
      acc0[r] *= alpha; acc1[r] *= alpha;
      p0[r] = e0; p1[r] = e1;
    }

    // C-layout (row = hh*8+r, col = l16) -> LDS row-major 16x32
    #pragma unroll
    for (int r = 0; r < 8; ++r) {
      pw[(hh * 8 + r) * 32 + l16]      = (f16)p0[r];
      pw[(hh * 8 + r) * 32 + 16 + l16] = (f16)p1[r];
    }
    __builtin_amdgcn_wave_barrier();
    asm volatile("s_wait_dscnt 0" ::: "memory");

    // P A-fragment: lane m=l16, kvlocal = hh*8+e / 16+hh*8+e
    v16h pa = cat16(*(const v8h*)&pw[l16 * 32 + hh * 8],
                    *(const v8h*)&pw[l16 * 32 + 16 + hh * 8]);
    // V B-fragments from LDS V^T tile: lane n(dv)=l16, kv = hh*8+e / 16+hh*8+e
    const f16* vl0 = &Vlds[buf][l16][hh * 8];
    const f16* vl1 = &Vlds[buf][16 + l16][hh * 8];
    acc0 = wmma_f16(pa, cat16(*(const v8h*)vl0, *(const v8h*)(vl0 + 16)), acc0);
    acc1 = wmma_f16(pa, cat16(*(const v8h*)vl1, *(const v8h*)(vl1 + 16)), acc1);

    if (wave == 0) asm volatile("s_wait_asynccnt 0" ::: "memory");
    __syncthreads();
  }

  #pragma unroll
  for (int r = 0; r < 8; ++r) {
    const float inv = 1.0f / rsum[r];
    const int q = q0 + hh * 8 + r;
    float* orow = out + ((size_t)(b * LQ + q)) * CH + h * DH;
    orow[l16]      = acc0[r] * inv;
    orow[16 + l16] = acc1[r] * inv;
  }
}

// ---------------------------------------------------------------------------
extern "C" void kernel_launch(void* const* d_in, const int* in_sizes, int n_in,
                              void* d_out, int out_size, void* d_ws, size_t ws_size,
                              hipStream_t stream) {
  (void)in_sizes; (void)n_in; (void)out_size; (void)ws_size;
  const float* hidden = (const float*)d_in[0];
  const float* inputs = (const float*)d_in[1];
  const float* ln1_g  = (const float*)d_in[2];
  const float* ln1_b  = (const float*)d_in[3];
  const float* ln2_g  = (const float*)d_in[4];
  const float* ln2_b  = (const float*)d_in[5];
  const float* Wq     = (const float*)d_in[6];
  const float* bq     = (const float*)d_in[7];
  const float* Wk     = (const float*)d_in[8];
  const float* bk     = (const float*)d_in[9];
  const float* Wv     = (const float*)d_in[10];
  const float* bv     = (const float*)d_in[11];
  float* out = (float*)d_out;

  char* ws = (char*)d_ws;
  size_t off = 0;
  auto carve = [&](size_t bytes) -> char* {
    char* p = ws + off;
    off += (bytes + 255) & ~(size_t)255;
    return p;
  };
  f16* hs_ln  = (f16*)carve((size_t)Bz * LQ  * QDIM  * 2);   //  4 MB
  f16* inp_ln = (f16*)carve((size_t)Bz * LKV * KVDIM * 2);   // 32 MB
  f16* Wqt    = (f16*)carve((size_t)CH * QDIM  * 2);
  f16* Wkt    = (f16*)carve((size_t)CH * KVDIM * 2);
  f16* Wvt    = (f16*)carve((size_t)CH * KVDIM * 2);
  f16* Qf     = (f16*)carve((size_t)Bz * LQ  * CH * 2);      //  1 MB
  f16* Kf     = (f16*)carve((size_t)Bz * LKV * CH * 2);      // 16 MB
  f16* Vtf    = (f16*)carve((size_t)Bz * CH * LKV * 2);      // 16 MB (transposed)

  // 1) LayerNorms -> f16
  ln_to_f16<QDIM> <<<Bz * LQ,  256, 0, stream>>>(hidden, ln1_g, ln1_b, hs_ln);
  ln_to_f16<KVDIM><<<Bz * LKV, 256, 0, stream>>>(inputs, ln2_g, ln2_b, inp_ln);

  // 2) Weight cast + transpose -> f16 [N][K]
  cast_transpose<<<(QDIM * CH + 255) / 256, 256, 0, stream>>>(Wq, Wqt, QDIM, CH);
  cast_transpose<<<(KVDIM * CH + 255) / 256, 256, 0, stream>>>(Wk, Wkt, KVDIM, CH);
  cast_transpose<<<(KVDIM * CH + 255) / 256, 256, 0, stream>>>(Wv, Wvt, KVDIM, CH);

  // 3) Projections (WMMA). Q pre-scaled by 1/sqrt(32).
  const float qscale = 0.17677669529663687f;
  dim3 gq((Bz * LQ) / 64, CH / 32);
  gemm_wmma_f16<<<gq, 256, 0, stream>>>(hs_ln, Wqt, bq, Qf,
                                        Bz * LQ, CH, QDIM, qscale, 0);
  dim3 gkv((Bz * LKV) / 64, CH / 32);
  gemm_wmma_f16<<<gkv, 256, 0, stream>>>(inp_ln, Wkt, bk, Kf,
                                         Bz * LKV, CH, KVDIM, 1.0f, 0);
  gemm_wmma_f16<<<gkv, 256, 0, stream>>>(inp_ln, Wvt, bv, Vtf,
                                         Bz * LKV, CH, KVDIM, 1.0f, 1);

  // 4) Flash attention (WMMA), f32 output
  dim3 ga(Bz * H, LQ / 128);
  flash_attn_wmma<<<ga, 256, 0, stream>>>(Qf, Kf, Vtf, out);
}
